// CausalSelfAttention_50517405335966
// MI455X (gfx1250) — matrix operand order repaired and checked by reading, not compiled, not run on hardware
//
#include <hip/hip_runtime.h>

// ---------------------------------------------------------------------------
// CausalSelfAttention forward for MI455X (gfx1250, wave32, WMMA bf16,
// async global->LDS tile staging, double-buffered).
// B=2, T=2048, C=1024, H=16, HD=64
// ---------------------------------------------------------------------------

#define B_  2
#define T_  2048
#define C_  1024
#define H_  16
#define HD_ 64
#define M_  (B_ * T_)      // 4096 rows of x
#define N3_ (3 * C_)       // 3072

typedef __attribute__((ext_vector_type(8)))  float  v8f;
typedef __attribute__((ext_vector_type(16))) __bf16 v16bf;

union BFrag {
    v16bf          v;
    unsigned int   q[8];
    unsigned short h[16];
};

__device__ __forceinline__ unsigned short f2bf(float x) {
    unsigned int u = __float_as_uint(x);
    unsigned int r = u + 0x7FFFu + ((u >> 16) & 1u);   // round-to-nearest-even
    return (unsigned short)(r >> 16);
}

__device__ __forceinline__ v8f wmma_bf16(v16bf a, v16bf b, v8f c) {
    return __builtin_amdgcn_wmma_f32_16x16x32_bf16(
        false, a, false, b, (short)0, c, false, false);
}

// --- CDNA5 async global->LDS copy (tracked by ASYNCcnt) --------------------
__device__ __forceinline__ unsigned lds_addr32(const void* p) {
    return (unsigned)(uintptr_t)p;       // low 32 bits of generic ptr = LDS offset
}

__device__ __forceinline__ void async_b128(unsigned lds_off, const void* gptr) {
    asm volatile("global_load_async_to_lds_b128 %0, %1, off"
                 :: "v"(lds_off), "v"((unsigned long long)(uintptr_t)gptr)
                 : "memory");
}

template <int N>
__device__ __forceinline__ void wait_asynccnt() {
    asm volatile("s_wait_asynccnt %0" :: "i"(N) : "memory");
}

// A-matrix fragment (16x32 bf16, row-major source, row stride ld in ushorts).
__device__ __forceinline__ v16bf load_a_frag(const unsigned short* p, int ld) {
    int lane = threadIdx.x & 31;
    int m = lane & 15, half = lane >> 4;
    const unsigned short* r = p + m * ld;
    BFrag f;
#pragma unroll
    for (int i = 0; i < 8; ++i) {
        int k = ((i & 4) ? 16 : 0) + 8 * half + 2 * (i & 3);
        f.q[i] = *(const unsigned int*)(r + k);
    }
    return f.v;
}

// B-matrix fragment (32x16 bf16, row-major [K][N] source, row stride ld).
__device__ __forceinline__ v16bf load_b_frag(const unsigned short* p, int ld) {
    int lane = threadIdx.x & 31;
    int n = lane & 15, half = lane >> 4;
    BFrag f;
#pragma unroll
    for (int s = 0; s < 16; ++s) {
        int k = ((s & 8) ? 16 : 0) + 8 * half + (s & 7);
        f.h[s] = p[k * ld + n];
    }
    return f.v;
}

// ---------------------------------------------------------------------------
// Kernel 1: fp32 -> bf16 conversion
// ---------------------------------------------------------------------------
__global__ void k_conv_bf16(const float* __restrict__ src,
                            unsigned short* __restrict__ dst, int n) {
    int i = blockIdx.x * blockDim.x + threadIdx.x;
    if (i < n) dst[i] = f2bf(src[i]);
}

// ---------------------------------------------------------------------------
// GEMM tiling shared by kernels 2 & 4:
//   block tile 128(M) x 256(N), 8 waves (2x4), wave tile 64x64 (4x4 WMMA).
//   K staged in 32-deep bf16 tiles, double-buffered via async->LDS.
// ---------------------------------------------------------------------------
#define LDA_ 40
#define LDB_ 264

struct GemmSmem {
    unsigned short As[2][128 * LDA_];
    unsigned short Bs[2][32 * LDB_];
};

template <int LDG_B>
__device__ __forceinline__ void gemm_issue_stage(
    GemmSmem& sm, int sel,
    const unsigned short* __restrict__ Ag,   // &A[M0][k0]
    const unsigned short* __restrict__ Bg) { // &B[k0][N0]
    const int tid = threadIdx.x;
    {   // A tile: 128 x 32, 2 threads/row, 16 ushorts each
        const int row = tid >> 1, col = (tid & 1) * 16;
        const unsigned short* g = Ag + (size_t)row * C_ + col;
        unsigned d = lds_addr32(&sm.As[sel][row * LDA_ + col]);
        async_b128(d,      g);
        async_b128(d + 16, g + 8);
    }
    {   // B tile: 32 x 256, 8 threads/row, 32 ushorts each
        const int row = tid >> 3, col = (tid & 7) * 32;
        const unsigned short* g = Bg + (size_t)row * LDG_B + col;
        unsigned d = lds_addr32(&sm.Bs[sel][row * LDB_ + col]);
        async_b128(d,      g);
        async_b128(d + 16, g + 8);
        async_b128(d + 32, g + 16);
        async_b128(d + 48, g + 24);
    }
}

// ---------------------------------------------------------------------------
// Kernel 2: QKV GEMM  qkv = x(bf16)[4096,1024] @ W_attn(bf16)[1024,3072] + b
// Epilogue scatters Q,V -> [B,H,T,HD], K -> [B,H,HD,T] (pre-transposed).
// ---------------------------------------------------------------------------
__global__ __launch_bounds__(256) void k_qkv_gemm(
    const unsigned short* __restrict__ xb,
    const unsigned short* __restrict__ Wab,
    const float* __restrict__ b_attn,
    unsigned short* __restrict__ Qh,
    unsigned short* __restrict__ Kt,
    unsigned short* __restrict__ Vh) {

    __shared__ __align__(16) GemmSmem sm;

    const int tid  = threadIdx.x;
    const int lane = tid & 31;
    const int wid  = tid >> 5;
    const int wm   = wid >> 2;          // 0..1
    const int wn   = wid & 3;           // 0..3
    const int M0   = blockIdx.y * 128;
    const int N0   = blockIdx.x * 256;

    v8f acc[4][4];
#pragma unroll
    for (int i = 0; i < 4; ++i)
#pragma unroll
        for (int j = 0; j < 4; ++j)
            acc[i][j] = (v8f){0.f, 0.f, 0.f, 0.f, 0.f, 0.f, 0.f, 0.f};

    const unsigned short* Ag = xb  + (size_t)M0 * C_;
    const unsigned short* Bg = Wab + N0;

    gemm_issue_stage<N3_>(sm, 0, Ag, Bg);

    const int KT = C_ / 32;             // 32 k-steps
    for (int kt = 0; kt < KT; ++kt) {
        const int cur = kt & 1;
        if (kt + 1 < KT) {
            gemm_issue_stage<N3_>(sm, cur ^ 1,
                                  Ag + (kt + 1) * 32,
                                  Bg + (size_t)(kt + 1) * 32 * N3_);
            wait_asynccnt<6>();         // stage kt (6 ops) complete
        } else {
            wait_asynccnt<0>();
        }
        __syncthreads();

        v16bf a[4], b[4];
#pragma unroll
        for (int i = 0; i < 4; ++i)
            a[i] = load_a_frag(&sm.As[cur][(wm * 64 + 16 * i) * LDA_], LDA_);
#pragma unroll
        for (int j = 0; j < 4; ++j)
            b[j] = load_b_frag(&sm.Bs[cur][wn * 64 + 16 * j], LDB_);
#pragma unroll
        for (int i = 0; i < 4; ++i)
#pragma unroll
            for (int j = 0; j < 4; ++j)
                acc[i][j] = wmma_bf16(a[i], b[j], acc[i][j]);
        __syncthreads();
    }

    // epilogue: bias + scatter into head layouts
    const int colL  = lane & 15;
    const int rbase = (lane >> 4) * 8;
#pragma unroll
    for (int i = 0; i < 4; ++i) {
#pragma unroll
        for (int j = 0; j < 4; ++j) {
            const int gn   = N0 + wn * 64 + j * 16 + colL;
            const float bi = b_attn[gn];
            const int sect = gn >> 10;          // 0=Q 1=K 2=V
            const int c    = gn & 1023;
            const int h    = c >> 6;
            const int d    = c & 63;
#pragma unroll
            for (int v = 0; v < 8; ++v) {
                const int gm = M0 + wm * 64 + i * 16 + rbase + v;
                const int bb = gm >> 11;
                const int t  = gm & 2047;
                const unsigned short bv = f2bf(acc[i][j][v] + bi);
                const size_t hbx = (size_t)(bb * H_ + h);
                if (sect == 0)      Qh[(hbx * T_ + t) * HD_ + d] = bv;
                else if (sect == 1) Kt[(hbx * HD_ + d) * T_ + t] = bv;
                else                Vh[(hbx * T_ + t) * HD_ + d] = bv;
            }
        }
    }
}

// ---------------------------------------------------------------------------
// Kernel 3: flash attention. grid = (T/64, B*H), 128 threads = 4 waves.
// Each wave owns 16 query rows; 64-key tiles, online softmax,
// K^T/V tiles double-buffered via async->LDS.
// ---------------------------------------------------------------------------
__global__ __launch_bounds__(128) void k_attn(
    const unsigned short* __restrict__ Qh,
    const unsigned short* __restrict__ Kt,
    const unsigned short* __restrict__ Vh,
    unsigned short* __restrict__ Yb) {

    __shared__ __align__(16) unsigned short KtS[2][64 * 72];  // [d][key]
    __shared__ __align__(16) unsigned short VS [2][64 * 72];  // [key][d]
    __shared__ float          Sbuf[4][16 * 68];
    __shared__ unsigned short Pbuf[4][16 * 72];
    __shared__ float          alphaBuf[4][16];
    __shared__ float          lBuf[4][16];

    const int tid  = threadIdx.x;
    const int lane = tid & 31;
    const int wid  = tid >> 5;                 // 0..3
    const int qb   = blockIdx.x;               // 0..31
    const int bh   = blockIdx.y;               // 0..31
    const int b    = bh >> 4, h = bh & 15;
    const size_t hb = (size_t)(b * H_ + h);

    const int ldrow = tid >> 1, ldc0 = (tid & 1) * 32;
    const unsigned short* KtBase = Kt + (hb * HD_ + ldrow) * T_ + ldc0;
    const unsigned short* VBase  = Vh + (hb * T_ + ldrow) * HD_ + ldc0;

    // stage issue: 8 async b128 per thread
    auto issue = [&](int kb, int sel) {
        const unsigned short* gk = KtBase + kb * 64;
        unsigned dk = lds_addr32(&KtS[sel][ldrow * 72 + ldc0]);
        const unsigned short* gv = VBase + (size_t)kb * 64 * HD_;
        unsigned dv = lds_addr32(&VS[sel][ldrow * 72 + ldc0]);
#pragma unroll
        for (int q = 0; q < 4; ++q) async_b128(dk + 16 * q, gk + 8 * q);
#pragma unroll
        for (int q = 0; q < 4; ++q) async_b128(dv + 16 * q, gv + 8 * q);
    };

    // Q A-fragments straight from global (row-major, stride HD)
    const unsigned short* Qbase = Qh + (hb * T_ + (size_t)qb * 64 + wid * 16) * HD_;
    v16bf aq[2];
#pragma unroll
    for (int kc = 0; kc < 2; ++kc) aq[kc] = load_a_frag(Qbase + 32 * kc, HD_);

    v8f o[4];
#pragma unroll
    for (int dt = 0; dt < 4; ++dt)
        o[dt] = (v8f){0.f, 0.f, 0.f, 0.f, 0.f, 0.f, 0.f, 0.f};
    float m_i = -INFINITY, l_i = 0.f;          // valid in lanes 0..15

    const int qrow = qb * 64 + wid * 16 + lane;   // query row for lane<16

    issue(0, 0);

    for (int kb = 0; kb <= qb; ++kb) {
        const int cur = kb & 1;
        if (kb < qb) {
            issue(kb + 1, cur ^ 1);
            wait_asynccnt<8>();
        } else {
            wait_asynccnt<0>();
        }
        __syncthreads();

        // S = Q K^T  (wave tile 16x64 = 4 N-tiles, K-depth 64 = 2 chunks)
#pragma unroll
        for (int j = 0; j < 4; ++j) {
            v8f s = (v8f){0.f, 0.f, 0.f, 0.f, 0.f, 0.f, 0.f, 0.f};
#pragma unroll
            for (int kc = 0; kc < 2; ++kc) {
                v16bf bf = load_b_frag(&KtS[cur][(32 * kc) * 72 + 16 * j], 72);
                s = wmma_bf16(aq[kc], bf, s);
            }
            const int col = (lane & 15) + 16 * j;
            const int rb  = (lane >> 4) * 8;
#pragma unroll
            for (int v = 0; v < 8; ++v)
                Sbuf[wid][(rb + v) * 68 + col] = s[v];
        }

        // online softmax: lanes 0..15, one query row each
        if (lane < 16) {
            const int r = lane;
            const int kmax = qrow - kb * 64;   // keys j<=kmax are causal-valid
            float mloc = -INFINITY;
#pragma unroll 8
            for (int j = 0; j < 64; ++j)
                if (j <= kmax)
                    mloc = fmaxf(mloc, Sbuf[wid][r * 68 + j] * 0.125f);
            const float mnew  = fmaxf(m_i, mloc);
            const float alpha = (m_i == -INFINITY) ? 0.f : __expf(m_i - mnew);
            float lsum = 0.f;
#pragma unroll 8
            for (int j = 0; j < 64; ++j) {
                float p = 0.f;
                if (j <= kmax)
                    p = __expf(Sbuf[wid][r * 68 + j] * 0.125f - mnew);
                Pbuf[wid][r * 72 + j] = f2bf(p);
                lsum += p;
            }
            l_i = l_i * alpha + lsum;
            m_i = mnew;
            alphaBuf[wid][r] = alpha;
        }

        // rescale O by alpha, then O += P V
        const int rb2 = (lane >> 4) * 8;
#pragma unroll
        for (int dt = 0; dt < 4; ++dt)
#pragma unroll
            for (int v = 0; v < 8; ++v)
                o[dt][v] *= alphaBuf[wid][rb2 + v];

        v16bf ap[2];
#pragma unroll
        for (int kc = 0; kc < 2; ++kc)
            ap[kc] = load_a_frag(&Pbuf[wid][32 * kc], 72);
#pragma unroll
        for (int dt = 0; dt < 4; ++dt)
#pragma unroll
            for (int kc = 0; kc < 2; ++kc) {
                v16bf bv = load_b_frag(&VS[cur][(32 * kc) * 72 + 16 * dt], 72);
                o[dt] = wmma_bf16(ap[kc], bv, o[dt]);
            }
        __syncthreads();
    }

    if (lane < 16) lBuf[wid][lane] = l_i;

    const int col = lane & 15;
    const int rb  = (lane >> 4) * 8;
#pragma unroll
    for (int dt = 0; dt < 4; ++dt)
#pragma unroll
        for (int v = 0; v < 8; ++v) {
            const int r    = rb + v;
            const float iv = 1.0f / lBuf[wid][r];
            const int trow = qb * 64 + wid * 16 + r;
            Yb[((size_t)b * T_ + trow) * C_ + h * HD_ + dt * 16 + col] =
                f2bf(o[dt][v] * iv);
        }
}

// ---------------------------------------------------------------------------
// Kernel 4: proj GEMM  out = Y(bf16)[4096,1024] @ W_proj(bf16)[1024,1024] + b
// ---------------------------------------------------------------------------
__global__ __launch_bounds__(256) void k_proj_gemm(
    const unsigned short* __restrict__ Yb,
    const unsigned short* __restrict__ Wpb,
    const float* __restrict__ b_proj,
    float* __restrict__ out) {

    __shared__ __align__(16) GemmSmem sm;

    const int tid  = threadIdx.x;
    const int lane = tid & 31;
    const int wid  = tid >> 5;
    const int wm   = wid >> 2;
    const int wn   = wid & 3;
    const int M0   = blockIdx.y * 128;
    const int N0   = blockIdx.x * 256;

    v8f acc[4][4];
#pragma unroll
    for (int i = 0; i < 4; ++i)
#pragma unroll
        for (int j = 0; j < 4; ++j)
            acc[i][j] = (v8f){0.f, 0.f, 0.f, 0.f, 0.f, 0.f, 0.f, 0.f};

    const unsigned short* Ag = Yb  + (size_t)M0 * C_;
    const unsigned short* Bg = Wpb + N0;

    gemm_issue_stage<C_>(sm, 0, Ag, Bg);

    const int KT = C_ / 32;
    for (int kt = 0; kt < KT; ++kt) {
        const int cur = kt & 1;
        if (kt + 1 < KT) {
            gemm_issue_stage<C_>(sm, cur ^ 1,
                                 Ag + (kt + 1) * 32,
                                 Bg + (size_t)(kt + 1) * 32 * C_);
            wait_asynccnt<6>();
        } else {
            wait_asynccnt<0>();
        }
        __syncthreads();

        v16bf a[4], b[4];
#pragma unroll
        for (int i = 0; i < 4; ++i)
            a[i] = load_a_frag(&sm.As[cur][(wm * 64 + 16 * i) * LDA_], LDA_);
#pragma unroll
        for (int j = 0; j < 4; ++j)
            b[j] = load_b_frag(&sm.Bs[cur][wn * 64 + 16 * j], LDB_);
#pragma unroll
        for (int i = 0; i < 4; ++i)
#pragma unroll
            for (int j = 0; j < 4; ++j)
                acc[i][j] = wmma_bf16(a[i], b[j], acc[i][j]);
        __syncthreads();
    }

    const int colL  = lane & 15;
    const int rbase = (lane >> 4) * 8;
#pragma unroll
    for (int i = 0; i < 4; ++i)
#pragma unroll
        for (int j = 0; j < 4; ++j) {
            const int gn   = N0 + wn * 64 + j * 16 + colL;
            const float bi = b_proj[gn];
#pragma unroll
            for (int v = 0; v < 8; ++v) {
                const int gm = M0 + wm * 64 + i * 16 + rbase + v;
                out[(size_t)gm * C_ + gn] = acc[i][j][v] + bi;
            }
        }
}

// ---------------------------------------------------------------------------
// Host entry
// ---------------------------------------------------------------------------
extern "C" void kernel_launch(void* const* d_in, const int* in_sizes, int n_in,
                              void* d_out, int out_size, void* d_ws, size_t ws_size,
                              hipStream_t stream) {
    const float* x      = (const float*)d_in[0];
    const float* W_attn = (const float*)d_in[1];
    const float* b_attn = (const float*)d_in[2];
    const float* W_proj = (const float*)d_in[3];
    const float* b_proj = (const float*)d_in[4];
    float* out = (float*)d_out;

    unsigned short* xb  = (unsigned short*)d_ws;          // 4096x1024
    unsigned short* Wab = xb  + (size_t)M_ * C_;          // 1024x3072
    unsigned short* Wpb = Wab + (size_t)C_ * N3_;         // 1024x1024
    unsigned short* Qh  = Wpb + (size_t)C_ * C_;          // [B,H,T,HD]
    unsigned short* Kt  = Qh  + (size_t)M_ * C_;          // [B,H,HD,T]
    unsigned short* Vh  = Kt  + (size_t)M_ * C_;          // [B,H,T,HD]
    unsigned short* Yb  = Vh  + (size_t)M_ * C_;          // 4096x1024

    const int nX  = M_ * C_;
    const int nWa = C_ * N3_;
    const int nWp = C_ * C_;

    k_conv_bf16<<<(nX  + 255) / 256, 256, 0, stream>>>(x,      xb,  nX);
    k_conv_bf16<<<(nWa + 255) / 256, 256, 0, stream>>>(W_attn, Wab, nWa);
    k_conv_bf16<<<(nWp + 255) / 256, 256, 0, stream>>>(W_proj, Wpb, nWp);

    k_qkv_gemm<<<dim3(N3_ / 256, M_ / 128), 256, 0, stream>>>(
        xb, Wab, b_attn, Qh, Kt, Vh);

    k_attn<<<dim3(T_ / 64, B_ * H_), 128, 0, stream>>>(Qh, Kt, Vh, Yb);

    k_proj_gemm<<<dim3(C_ / 256, M_ / 128), 256, 0, stream>>>(
        Yb, Wpb, b_proj, out);
}
